// Attention_43636867728088
// MI455X (gfx1250) — matrix operand order repaired and checked by reading, not compiled, not run on hardware
//
#include <hip/hip_runtime.h>
#include <hip/hip_bf16.h>
#include <math.h>

// ---------------- constants ----------------
#define BB     4
#define DIM    192
#define HEADS  6
#define CC     32          // channels per head
#define HH     256
#define WW     256
#define HW     65536       // 256*256
#define QKVC   576         // 3*DIM
#define NTILES 16384       // BB*HW/16
#define KSPLIT 32          // K-partitions per (b,h) in attn

typedef __attribute__((ext_vector_type(16))) __bf16 v16bf;
typedef __attribute__((ext_vector_type(8)))  float  v8f;

static __device__ __forceinline__ v8f wmma_bf16(v16bf a, v16bf b, v8f c) {
  return __builtin_amdgcn_wmma_f32_16x16x32_bf16(false, a, false, b, (short)0, c,
                                                 false, false);
}

// Assemble an A-matrix fragment from 16 contiguous floats
// (k = kb..kb+15 in elements 0..15).
static __device__ __forceinline__ v16bf load_a_frag(const float* __restrict__ p) {
  float4 r0 = *(const float4*)(p);
  float4 r1 = *(const float4*)(p + 4);
  float4 r2 = *(const float4*)(p + 8);
  float4 r3 = *(const float4*)(p + 12);
  v16bf a;
  a[0]  = (__bf16)r0.x; a[1]  = (__bf16)r0.y; a[2]  = (__bf16)r0.z; a[3]  = (__bf16)r0.w;
  a[4]  = (__bf16)r1.x; a[5]  = (__bf16)r1.y; a[6]  = (__bf16)r1.z; a[7]  = (__bf16)r1.w;
  a[8]  = (__bf16)r2.x; a[9]  = (__bf16)r2.y; a[10] = (__bf16)r2.z; a[11] = (__bf16)r2.w;
  a[12] = (__bf16)r3.x; a[13] = (__bf16)r3.y; a[14] = (__bf16)r3.z; a[15] = (__bf16)r3.w;
  return a;
}

// Assemble a B-matrix fragment from 16 contiguous floats (k = kb..kb+15).
static __device__ __forceinline__ v16bf load_b_frag16(const float* __restrict__ p) {
  float4 r0 = *(const float4*)(p);
  float4 r1 = *(const float4*)(p + 4);
  float4 r2 = *(const float4*)(p + 8);
  float4 r3 = *(const float4*)(p + 12);
  v16bf b;
  b[0]  = (__bf16)r0.x; b[1]  = (__bf16)r0.y; b[2]  = (__bf16)r0.z; b[3]  = (__bf16)r0.w;
  b[4]  = (__bf16)r1.x; b[5]  = (__bf16)r1.y; b[6]  = (__bf16)r1.z; b[7]  = (__bf16)r1.w;
  b[8]  = (__bf16)r2.x; b[9]  = (__bf16)r2.y; b[10] = (__bf16)r2.z; b[11] = (__bf16)r2.w;
  b[12] = (__bf16)r3.x; b[13] = (__bf16)r3.y; b[14] = (__bf16)r3.z; b[15] = (__bf16)r3.w;
  return b;
}

// ---------------- small helper kernels ----------------

// L2-normalize contiguous rows of length 256 (over the W axis).
__global__ void rownorm256_kernel(float* __restrict__ dst,
                                  const float* __restrict__ src) {
  int row = blockIdx.x;
  int t   = threadIdx.x;            // 256 threads
  __shared__ float red[256];
  float v = src[(size_t)row * 256 + t];
  red[t] = v * v;
  __syncthreads();
  for (int s = 128; s > 0; s >>= 1) {
    if (t < s) red[t] += red[t + s];
    __syncthreads();
  }
  float scale = 1.0f / fmaxf(sqrtf(red[0]), 1e-12f);
  dst[(size_t)row * 256 + t] = v * scale;
}

// L2-normalize rows of length HW (over the flattened spatial axis), in place.
__global__ void headnorm_kernel(float* __restrict__ data) {
  int row = blockIdx.x;
  int t   = threadIdx.x;            // 256 threads
  float* p = data + (size_t)row * HW;
  __shared__ float red[256];
  float s = 0.f;
  for (int i = t; i < HW; i += 256) { float v = p[i]; s += v * v; }
  red[t] = s;
  __syncthreads();
  for (int st = 128; st > 0; st >>= 1) {
    if (t < st) red[t] += red[t + st];
    __syncthreads();
  }
  float scale = 1.0f / fmaxf(sqrtf(red[0]), 1e-12f);
  for (int i = t; i < HW; i += 256) p[i] *= scale;
}

// 3x3 SAME conv, 3 -> 192 channels (superpixel projection).
__global__ void projconv_kernel(const float* __restrict__ spn,
                                const float* __restrict__ pw,
                                float* __restrict__ spm) {
  size_t idx = (size_t)blockIdx.x * 256 + threadIdx.x;  // DIM*HW threads
  int p  = (int)(idx & (HW - 1));
  int co = (int)(idx >> 16);
  int y = p >> 8, x = p & 255;
  float acc = 0.f;
  for (int ci = 0; ci < 3; ci++) {
    const float* s = spn + (size_t)ci * HW;
    const float* w = pw + ((size_t)co * 3 + ci) * 9;
    #pragma unroll
    for (int dy = -1; dy <= 1; dy++)
      #pragma unroll
      for (int dx = -1; dx <= 1; dx++) {
        int yy = y + dy, xx = x + dx;
        if (yy >= 0 && yy < HH && xx >= 0 && xx < WW)
          acc += w[(dy + 1) * 3 + (dx + 1)] * s[yy * WW + xx];
      }
  }
  spm[idx] = acc;
}

// Depthwise 3x3 SAME conv over the 576 qkv channels; scatter into q/k/v.
__global__ void dwconv_kernel(const float* __restrict__ qkv,
                              const float* __restrict__ dww,
                              float* __restrict__ q, float* __restrict__ k,
                              float* __restrict__ v) {
  size_t idx = (size_t)blockIdx.x * 256 + threadIdx.x;  // BB*QKVC*HW
  int p = (int)(idx & (HW - 1));
  size_t bc = idx >> 16;
  int ch = (int)(bc % QKVC);
  int b  = (int)(bc / QKVC);
  int y = p >> 8, x = p & 255;
  const float* src = qkv + (bc << 16);
  const float* wgt = dww + (size_t)ch * 9;
  float acc = 0.f;
  #pragma unroll
  for (int dy = -1; dy <= 1; dy++)
    #pragma unroll
    for (int dx = -1; dx <= 1; dx++) {
      int yy = y + dy, xx = x + dx;
      if (yy >= 0 && yy < HH && xx >= 0 && xx < WW)
        acc += wgt[(dy + 1) * 3 + (dx + 1)] * src[yy * WW + xx];
    }
  int part = ch / DIM, cc = ch % DIM;
  float* dst = (part == 0) ? q : (part == 1) ? k : v;
  dst[((size_t)b * DIM + cc) * HW + p] = acc;
}

// q[b,c,p] *= sp_mean[c,p]   (sp_mean broadcast over batch)
__global__ void mulspm_kernel(float* __restrict__ q,
                              const float* __restrict__ spm) {
  size_t idx = (size_t)blockIdx.x * 256 + threadIdx.x;  // BB*DIM*HW
  size_t rem = idx % ((size_t)DIM * HW);
  q[idx] *= spm[rem];
}

// ---------------- WMMA GEMM kernels ----------------

// 1x1 conv as GEMM: out[b, m, p] = sum_k wgt[m,k] * in[b,k,p]
// One wave = 1 M-tile x 4 N-tiles (A fragment shared across 4 WMMAs per K-step).
__global__ void conv1x1_wmma_kernel(const float* __restrict__ in,
                                    const float* __restrict__ wgt,
                                    float* __restrict__ out, int M, int K) {
  int wave = threadIdx.x >> 5;
  int lane = threadIdx.x & 31;
  const int NG = NTILES / 4;          // 4096 groups of 4 N-tiles
  size_t g = (size_t)blockIdx.x * 8 + wave;
  int mt = (int)(g / NG);
  int ng = (int)(g % NG);
  int b     = ng >> 10;               // 1024 groups per batch
  int pbase = (ng & 1023) << 6;       // 64 pixels per group
  int m   = lane & 15;
  int abk = (lane < 16) ? 0 : 16;     // A k-base
  int bko = (lane < 16) ? 0 : 16;     // B k-offset
  int nc  = lane & 15;                // column within N-tile
  v8f acc[4];
  #pragma unroll
  for (int t = 0; t < 4; t++) acc[t] = (v8f){0.f,0.f,0.f,0.f,0.f,0.f,0.f,0.f};
  for (int kk = 0; kk < K; kk += 32) {
    v16bf a = load_a_frag(wgt + (size_t)(mt * 16 + m) * K + kk + abk);
    const float* bp = in + ((size_t)b * K + kk + bko) * HW + pbase + nc;
    if (kk + 32 < K) {  // uniform branch: prefetch next K-chunk of B
      __builtin_prefetch(bp + (size_t)32 * HW, 0, 1);
      __builtin_prefetch(bp + (size_t)40 * HW, 0, 1);
    }
    #pragma unroll
    for (int t = 0; t < 4; t++) {
      v16bf bm;
      #pragma unroll
      for (int e = 0; e < 16; e++) bm[e] = (__bf16)bp[(size_t)e * HW + t * 16];
      acc[t] = wmma_bf16(a, bm, acc[t]);
    }
  }
  int mr = (lane < 16) ? 0 : 8;
  #pragma unroll
  for (int t = 0; t < 4; t++)
    #pragma unroll
    for (int r = 0; r < 8; r++)
      out[((size_t)b * M + mt * 16 + r + mr) * HW + pbase + t * 16 + nc] = acc[t][r];
}

// partial[bh, blk, c, d] = sum_{n in blk's K-slice} qs[bh,c,n]*kh[bh,d,n]
// Grid: (b,h) x KSPLIT blocks; 8 waves per block further partition the slice.
__global__ void attn_wmma_kernel(const float* __restrict__ qs,
                                 const float* __restrict__ kh,
                                 float* __restrict__ partial) {
  int bh  = blockIdx.x >> 5;          // KSPLIT = 32
  int blk = blockIdx.x & 31;
  int b = bh / HEADS, hd = bh % HEADS;
  int wave = threadIdx.x >> 5;
  int lane = threadIdx.x & 31;
  const float* qb = qs + ((size_t)b * DIM + hd * CC) * HW;
  const float* kb = kh + ((size_t)b * DIM + hd * CC) * HW;
  v8f acc0 = {0.f,0.f,0.f,0.f,0.f,0.f,0.f,0.f};
  v8f acc1 = acc0, acc2 = acc0, acc3 = acc0;
  int m   = lane & 15;
  int abk = (lane < 16) ? 0 : 16;
  int bko = (lane < 16) ? 0 : 16;
  int d   = lane & 15;
  const int kbase = blk * (HW / KSPLIT);           // 2048-wide slice
  const int kend  = kbase + (HW / KSPLIT);
  for (int ck = kbase + wave * 32; ck < kend; ck += 256) {
    v16bf a0 = load_a_frag(qb + (size_t)m * HW + ck + abk);
    v16bf a1 = load_a_frag(qb + (size_t)(16 + m) * HW + ck + abk);
    v16bf b0 = load_b_frag16(kb + (size_t)d * HW + ck + bko);
    v16bf b1 = load_b_frag16(kb + (size_t)(16 + d) * HW + ck + bko);
    acc0 = wmma_bf16(a0, b0, acc0);
    acc1 = wmma_bf16(a0, b1, acc1);
    acc2 = wmma_bf16(a1, b0, acc2);
    acc3 = wmma_bf16(a1, b1, acc3);
  }
  // fixed-order cross-wave reduction in LDS
  __shared__ float red[32 * 32 * 8];  // [f][lane][wave], 32 KB
  #pragma unroll
  for (int r = 0; r < 8; r++) {
    red[((0 * 8 + r) * 32 + lane) * 8 + wave] = acc0[r];
    red[((1 * 8 + r) * 32 + lane) * 8 + wave] = acc1[r];
    red[((2 * 8 + r) * 32 + lane) * 8 + wave] = acc2[r];
    red[((3 * 8 + r) * 32 + lane) * 8 + wave] = acc3[r];
  }
  __syncthreads();
  #pragma unroll
  for (int j = 0; j < 4; j++) {
    int v  = threadIdx.x * 4 + j;   // 0..1023
    int f  = v >> 5;
    int ln = v & 31;
    float s = 0.f;
    #pragma unroll
    for (int w2 = 0; w2 < 8; w2++) s += red[(f * 32 + ln) * 8 + w2];
    int t2 = f >> 3, r = f & 7;
    int mi = t2 >> 1, ni = t2 & 1;
    int c  = mi * 16 + r + ((ln < 16) ? 0 : 8);
    int dd = ni * 16 + (ln & 15);
    partial[((size_t)bh * KSPLIT + blk) * (CC * CC) + c * CC + dd] = s;
  }
}

// attn[bh,cd] = temperature[h] * sum_blk partial[bh,blk,cd]   (fixed order)
__global__ void attn_reduce_kernel(const float* __restrict__ partial,
                                   const float* __restrict__ temp,
                                   float* __restrict__ attn_g) {
  int bh = blockIdx.x;
  float tmul = temp[bh % HEADS];
  #pragma unroll
  for (int j = 0; j < 4; j++) {
    int cd = threadIdx.x * 4 + j;   // 0..1023
    float s = 0.f;
    for (int blk = 0; blk < KSPLIT; blk++)
      s += partial[((size_t)bh * KSPLIT + blk) * (CC * CC) + cd];
    attn_g[(size_t)bh * (CC * CC) + cd] = s * tmul;
  }
}

// top-k soften + 4 softmaxes + relu branch, folded into one combined matrix:
// comb = wm0*(sum scales)*relu(a) + wm1*sum_i scales[i]*softmax(soften_i(a))
__global__ void softmix_kernel(const float* __restrict__ attn_g,
                               const float* __restrict__ scales,
                               const float* __restrict__ wmix,
                               float* __restrict__ comb) {
  int bh = blockIdx.x;
  int c  = threadIdx.x;  // 32 threads, one row each
  const float* row = attn_g + (size_t)bh * (CC * CC) + c * CC;
  float a[CC], s[CC];
  for (int i = 0; i < CC; i++) { a[i] = row[i]; s[i] = a[i]; }
  // insertion sort, descending
  for (int i = 1; i < CC; i++) {
    float key = s[i];
    int j = i - 1;
    while (j >= 0 && s[j] < key) { s[j + 1] = s[j]; j--; }
    s[j + 1] = key;
  }
  const int kks[4] = {CC / 2, (CC * 2) / 3, (CC * 3) / 4, (CC * 4) / 5};
  float w0 = wmix[0], w1 = wmix[1];
  float wmx = fmaxf(w0, w1);
  float e0 = expf(w0 - wmx), e1 = expf(w1 - wmx);
  float wm0 = e0 / (e0 + e1), wm1 = e1 / (e0 + e1);
  float out[CC];
  for (int i = 0; i < CC; i++) out[i] = 0.f;
  float sc_sum = 0.f;
  for (int t = 0; t < 4; t++) {
    float thr = s[kks[t] - 1];        // kk-th largest
    float sc = scales[t];
    sc_sum += sc;
    float sm[CC];
    float rmax = -3.4e38f;
    for (int i = 0; i < CC; i++) {
      sm[i] = (a[i] >= thr) ? a[i] : a[i] * 1e-6f;
      rmax = fmaxf(rmax, sm[i]);
    }
    float ssum = 0.f;
    for (int i = 0; i < CC; i++) { sm[i] = expf(sm[i] - rmax); ssum += sm[i]; }
    float inv = sc * wm1 / ssum;
    for (int i = 0; i < CC; i++) out[i] += inv * sm[i];
  }
  float w0s = wm0 * sc_sum;
  float* orow = comb + (size_t)bh * (CC * CC) + c * CC;
  for (int i = 0; i < CC; i++) orow[i] = out[i] + w0s * fmaxf(a[i], 0.f);
}

// out[bh, c, n] = sum_d comb[bh,c,d] * v[bh,d,n]  (M=32, K=32 -> one WMMA step)
__global__ void outgemm_wmma_kernel(const float* __restrict__ comb,
                                    const float* __restrict__ v,
                                    float* __restrict__ out) {
  int bh    = blockIdx.x >> 9;       // 512 chunks per (b,h)
  int chunk = blockIdx.x & 511;
  int b = bh / HEADS, hd = bh % HEADS;
  int wave = threadIdx.x >> 5;
  int lane = threadIdx.x & 31;
  int pbase = chunk * 128 + wave * 16;
  int m   = lane & 15;
  int abk = (lane < 16) ? 0 : 16;
  int bko = (lane < 16) ? 0 : 16;
  int n   = pbase + (lane & 15);
  v16bf a0 = load_a_frag(comb + (size_t)bh * (CC * CC) + m * CC + abk);
  v16bf a1 = load_a_frag(comb + (size_t)bh * (CC * CC) + (16 + m) * CC + abk);
  v16bf bm;
  const float* vp = v + ((size_t)b * DIM + hd * CC + bko) * HW + n;
  #pragma unroll
  for (int e = 0; e < 16; e++) bm[e] = (__bf16)vp[(size_t)e * HW];
  v8f z = {0.f,0.f,0.f,0.f,0.f,0.f,0.f,0.f};
  v8f d0 = wmma_bf16(a0, bm, z);
  v8f d1 = wmma_bf16(a1, bm, z);
  int mr = (lane < 16) ? 0 : 8;
  #pragma unroll
  for (int r = 0; r < 8; r++) {
    out[((size_t)b * DIM + hd * CC +      r + mr) * HW + n] = d0[r];
    out[((size_t)b * DIM + hd * CC + 16 + r + mr) * HW + n] = d1[r];
  }
}

// ---------------- launcher ----------------

extern "C" void kernel_launch(void* const* d_in, const int* in_sizes, int n_in,
                              void* d_out, int out_size, void* d_ws, size_t ws_size,
                              hipStream_t stream) {
  const float* x      = (const float*)d_in[0];   // [4,192,256,256]
  const float* sp     = (const float*)d_in[1];   // [1,3,256,256]
  const float* qkv_w  = (const float*)d_in[2];   // [576,192]
  const float* dw_w   = (const float*)d_in[3];   // [576,9]
  const float* proj_w = (const float*)d_in[4];   // [192,3,3,3]
  const float* out_w  = (const float*)d_in[5];   // [192,192]
  const float* temp   = (const float*)d_in[6];   // [6]
  const float* ascl   = (const float*)d_in[7];   // [4]
  const float* wmix   = (const float*)d_in[8];   // [2]
  float* outp = (float*)d_out;                   // [4,192,256,256]

  // workspace layout (floats)
  float* ws = (float*)d_ws;
  const size_t QKV_SZ = (size_t)BB * QKVC * HW;      // 150,994,944
  const size_t CH_SZ  = (size_t)BB * DIM * HW;       //  50,331,648
  float* qkv  = ws;                                  // also reused as outbuf
  float* qb   = qkv + QKV_SZ;
  float* kb   = qb + CH_SZ;
  float* vb   = kb + CH_SZ;
  float* spn  = vb + CH_SZ;                          // 3*HW
  float* spm  = spn + (size_t)3 * HW;                // DIM*HW
  float* attn = spm + (size_t)DIM * HW;              // 24*32*32
  float* comb = attn + (size_t)BB * HEADS * CC * CC;
  float* part = comb + (size_t)BB * HEADS * CC * CC; // 24*KSPLIT*1024
  float* outbuf = qkv;                               // alias dead qkv region

  // 1) superpixel row-norm + projection conv -> sp_mean (batch-invariant)
  rownorm256_kernel<<<3 * HH, 256, 0, stream>>>(spn, sp);
  projconv_kernel<<<(DIM * HW) / 256, 256, 0, stream>>>(spn, proj_w, spm);

  // 2) qkv 1x1 conv (bf16 WMMA GEMM): M=576, K=192
  conv1x1_wmma_kernel<<<(QKVC / 16) * (NTILES / 4) / 8, 256, 0, stream>>>(
      x, qkv_w, qkv, QKVC, DIM);

  // 3) depthwise 3x3 -> q, k, v
  dwconv_kernel<<<(unsigned)(QKV_SZ / 256), 256, 0, stream>>>(qkv, dw_w, qb, kb, vb);

  // 4) q: L2 over W, * sp_mean, L2 over hw (per head-channel); k: L2 over hw
  rownorm256_kernel<<<BB * DIM * HH, 256, 0, stream>>>(qb, qb);
  mulspm_kernel<<<(unsigned)(CH_SZ / 256), 256, 0, stream>>>(qb, spm);
  headnorm_kernel<<<BB * DIM, 256, 0, stream>>>(qb);
  headnorm_kernel<<<BB * DIM, 256, 0, stream>>>(kb);

  // 5) attn = qs . kh^T (bf16 WMMA, K split 32-way per head for occupancy)
  attn_wmma_kernel<<<BB * HEADS * KSPLIT, 256, 0, stream>>>(qb, kb, part);
  attn_reduce_kernel<<<BB * HEADS, 256, 0, stream>>>(part, temp, attn);

  // 6) fold topk-soften/softmax/relu branches into one combined 32x32
  softmix_kernel<<<BB * HEADS, 32, 0, stream>>>(attn, ascl, wmix, comb);

  // 7) out = combined @ vh (bf16 WMMA, single K step)
  outgemm_wmma_kernel<<<BB * HEADS * 512, 256, 0, stream>>>(comb, vb, outbuf);

  // 8) final 1x1 conv (bf16 WMMA GEMM): M=192, K=192 -> d_out
  conv1x1_wmma_kernel<<<(DIM / 16) * (NTILES / 4) / 8, 256, 0, stream>>>(
      outbuf, out_w, outp, DIM, DIM);
}
